// VITSelfAttention_85564338471094
// MI455X (gfx1250) — matrix operand order repaired and checked by reading, not compiled, not run on hardware
//
#include <hip/hip_runtime.h>

// Problem constants (match reference)
#define B_  16
#define N_  4096
#define C_  512
#define D_SCALE 0.044194173824159216f   // 1/sqrt(512)

typedef __bf16 v16bf __attribute__((ext_vector_type(16)));
typedef __bf16 v8bf  __attribute__((ext_vector_type(8)));
typedef float  v8f   __attribute__((ext_vector_type(8)));

__device__ __forceinline__ unsigned short f2bf(float f) {
  unsigned u = __float_as_uint(f);
  u += 0x7fffu + ((u >> 16) & 1u);   // round-to-nearest-even
  return (unsigned short)(u >> 16);
}
__device__ __forceinline__ float bf2f(unsigned short h) {
  return __uint_as_float(((unsigned)h) << 16);
}

// ---------------------------------------------------------------------------
// Kernel 1: fp32 -> bf16 conversion (grid-stride)
// ---------------------------------------------------------------------------
__global__ void f2bf_kernel(const float* __restrict__ src,
                            unsigned short* __restrict__ dst, int n) {
  int i = blockIdx.x * blockDim.x + threadIdx.x;
  int stride = gridDim.x * blockDim.x;
  for (; i < n; i += stride) dst[i] = f2bf(src[i]);
}

// ---------------------------------------------------------------------------
// Kernel 2: projection  S[p][b][d][n] = sigmoid( sum_c W_p[d,c]*x[b,n,c] + b_p[d] )
//
// Workgroup = 8 waves. Each WG owns (p, b, 128-row d-slab, 64-col n-tile).
// The x n-tile (64 rows x 512 c bf16 = 64 KB, contiguous in memory) is staged
// into LDS once via gfx1250 async global->LDS copies, then all 8 waves (one
// 16-row d-tile each) stream B-fragments from LDS (ds_load_b128) into
// v_wmma_f32_16x16x32_bf16. A-fragments (W rows) come from global (L2-resident).
// ---------------------------------------------------------------------------
__global__ __launch_bounds__(256)
void proj_kernel(const unsigned short* __restrict__ xb,   // [B][N][C] bf16
                 const unsigned short* __restrict__ wb,   // [3][C][C] bf16
                 const float* __restrict__ bq,
                 const float* __restrict__ bk,
                 const float* __restrict__ bv,
                 unsigned short* __restrict__ sout) {     // [3][B][C][N] bf16
  __shared__ unsigned short xtile[64 * C_];               // 64 KB

  int blk = blockIdx.x;
  int ng   = blk & 63; blk >>= 6;   // 64 n-tiles of 64 columns
  int dblk = blk & 3;  blk >>= 2;   // 4 d-slabs of 128 rows
  int b    = blk & 15; blk >>= 4;   // batch
  int p    = blk;                    // 0=Q, 1=K, 2=V

  int tid = threadIdx.x;

  // ---- async stage of the x tile: rows [ng*64, ng*64+64) x all 512 c ----
  // Full rows are copied, so the 64 KB region is contiguous in global memory.
  {
    const unsigned short* xsrc = xb + ((size_t)b * N_ + (size_t)ng * 64) * C_;
    unsigned ldsbase = (unsigned)(uintptr_t)(&xtile[0]);
#pragma unroll
    for (int s = 0; s < 16; ++s) {               // 4096 x 16B chunks / 256 thr
      int q = tid + s * 256;
      unsigned goff = (unsigned)(q * 16);
      unsigned loff = ldsbase + (unsigned)(q * 16);
      asm volatile("global_load_async_to_lds_b128 %0, %1, %2"
                   :: "v"(loff), "v"(goff), "s"(xsrc)
                   : "memory");
    }
    asm volatile("s_wait_asynccnt 0x0" ::: "memory");
  }
  __syncthreads();

  // ---- WMMA: each wave computes a 16(d) x 64(n) strip ----
  int wave = tid >> 5;
  int lane = tid & 31;
  int half = lane >> 4;
  int lm   = lane & 15;
  int dt   = dblk * 8 + wave;                    // d-tile 0..31

  const __bf16* wrow =
      (const __bf16*)(wb + ((size_t)p * C_ + (size_t)(dt * 16 + lm)) * C_);
  const __bf16* lxr = (const __bf16*)(&xtile[(size_t)lm * C_]);

  v8f acc[4] = {};
  for (int kt = 0; kt < C_; kt += 32) {
    // A fragment: 16 contiguous k of the W row (k = kt + 16*half + s), matching B
    v8bf al = *(const v8bf*)(wrow + kt + 16 * half);
    v8bf ah = *(const v8bf*)(wrow + kt + 16 * half + 8);
    v16bf a;
#pragma unroll
    for (int i = 0; i < 8; ++i) { a[i] = al[i]; a[i + 8] = ah[i]; }

    int boff = kt + 16 * half;
#pragma unroll
    for (int t = 0; t < 4; ++t) {
      // B fragment: 16 contiguous bf16 of LDS row (n = ng*64 + t*16 + lm)
      v16bf bm = *(const v16bf*)(lxr + (size_t)t * 16 * C_ + boff);
      acc[t] = __builtin_amdgcn_wmma_f32_16x16x32_bf16(
          false, a, false, bm, (short)0, acc[t], false, false);
    }
  }

  const float* bias = (p == 0) ? bq : ((p == 1) ? bk : bv);
  size_t sbase = (((size_t)p * B_ + (size_t)b) * C_) * (size_t)N_;
  int nb = ng * 64 + lm;
#pragma unroll
  for (int t = 0; t < 4; ++t) {
    int n = nb + t * 16;
#pragma unroll
    for (int r = 0; r < 8; ++r) {
      int d = dt * 16 + r + 8 * half;
      float y = acc[t][r] + bias[d];
      y = 1.0f / (1.0f + __expf(-y));            // sigmoid
      sout[sbase + (size_t)d * N_ + n] = f2bf(y);
    }
  }
}

// ---------------------------------------------------------------------------
// Kernel 3: per-(b,c) pooled attention on 64x64 spatial tiles.
// One 256-thread workgroup per (b,c). 64 KB LDS with buffer reuse.
// Matmuls are register-tiled (2x4 / 4x4) to cut LDS traffic ~4-5x.
// ---------------------------------------------------------------------------
__global__ __launch_bounds__(256)
void attn_kernel(const unsigned short* __restrict__ sbf,  // [3][B][C][N] bf16
                 float* __restrict__ out) {               // [B][N][C] fp32
  __shared__ float lds[16384];       // 64 KB
  float* lQ  = lds;                  // 4096 : Q, later V
  float* lK  = lds + 4096;           // 4096 : K, later Kqv
  float* lkp = lds + 8192;           // 2048 : pooled k
  float* lqp = lds + 10240;          // 2048 : pooled q
  float* s1  = lds + 12288;          // 2048 : Qk  (64 x 32)
  float* s2  = lds + 14336;          // 2048 : Kq  (32 x 64)

  int bc  = blockIdx.x;              // b*C + c
  int b   = bc >> 9;
  int c   = bc & (C_ - 1);
  int tid = threadIdx.x;

  const unsigned short* Sq = sbf + ((size_t)0 * B_ * C_ + bc) * (size_t)N_;
  const unsigned short* Sk = sbf + ((size_t)1 * B_ * C_ + bc) * (size_t)N_;
  const unsigned short* Sv = sbf + ((size_t)2 * B_ * C_ + bc) * (size_t)N_;

  // 1) load Q, K
  for (int i = tid; i < 4096; i += 256) { lQ[i] = bf2f(Sq[i]); lK[i] = bf2f(Sk[i]); }
  __syncthreads();

  // 2) avg-pool over pw (kernel 2): qp,kp [32][64]
  for (int i = tid; i < 2048; i += 256) {
    int r = i >> 6, j = i & 63;
    lkp[i] = 0.5f * (lK[(2 * r) * 64 + j] + lK[(2 * r + 1) * 64 + j]);
    lqp[i] = 0.5f * (lQ[(2 * r) * 64 + j] + lQ[(2 * r + 1) * 64 + j]);
  }
  __syncthreads();

  // 3a) s1 = scale * Q @ kp^T (64x32): 2(qi) x 4(kj) register tile per thread
  {
    int qi0 = (tid >> 3) << 1;        // 32 row-pairs
    int kj0 = (tid & 7) << 2;         // 8 col quads
    float acc[2][4] = {};
    for (int m = 0; m < 64; ++m) {
      float q0 = lQ[qi0 * 64 + m];
      float q1 = lQ[(qi0 + 1) * 64 + m];
#pragma unroll
      for (int j = 0; j < 4; ++j) {
        float kv = lkp[(kj0 + j) * 64 + m];
        acc[0][j] += q0 * kv;
        acc[1][j] += q1 * kv;
      }
    }
#pragma unroll
    for (int r = 0; r < 2; ++r)
#pragma unroll
      for (int j = 0; j < 4; ++j)
        s1[(qi0 + r) * 32 + kj0 + j] = acc[r][j] * D_SCALE;
  }
  // 3b) s2 = scale * qp @ K^T (32x64): 2 x 4 register tile per thread
  {
    int qi0 = (tid >> 4) << 1;        // 16 row-pairs
    int kj0 = (tid & 15) << 2;        // 16 col quads
    float acc[2][4] = {};
    for (int m = 0; m < 64; ++m) {
      float q0 = lqp[qi0 * 64 + m];
      float q1 = lqp[(qi0 + 1) * 64 + m];
#pragma unroll
      for (int j = 0; j < 4; ++j) {
        float kv = lK[(kj0 + j) * 64 + m];
        acc[0][j] += q0 * kv;
        acc[1][j] += q1 * kv;
      }
    }
#pragma unroll
    for (int r = 0; r < 2; ++r)
#pragma unroll
      for (int j = 0; j < 4; ++j)
        s2[(qi0 + r) * 64 + kj0 + j] = acc[r][j] * D_SCALE;
  }
  __syncthreads();

  // 4) row softmaxes (s1: 64 rows of 32; s2: 32 rows of 64)
  if (tid < 64) {
    float* row = s1 + tid * 32;
    float mx = row[0];
    for (int j = 1; j < 32; ++j) mx = fmaxf(mx, row[j]);
    float s = 0.f;
    for (int j = 0; j < 32; ++j) { float e = __expf(row[j] - mx); row[j] = e; s += e; }
    float inv = 1.0f / s;
    for (int j = 0; j < 32; ++j) row[j] *= inv;
  } else if (tid < 96) {
    float* row = s2 + (tid - 64) * 64;
    float mx = row[0];
    for (int j = 1; j < 64; ++j) mx = fmaxf(mx, row[j]);
    float s = 0.f;
    for (int j = 0; j < 64; ++j) { float e = __expf(row[j] - mx); row[j] = e; s += e; }
    float inv = 1.0f / s;
    for (int j = 0; j < 64; ++j) row[j] *= inv;
  }
  __syncthreads();

  // 5) load V into Q's slot (Q no longer needed)
  for (int i = tid; i < 4096; i += 256) lQ[i] = bf2f(Sv[i]);
  __syncthreads();

  // 6) Kqv = s2 @ V (32x64) into K's slot: 2 x 4 register tile
  {
    int r0 = (tid >> 4) << 1;         // 16 row-pairs
    int j0 = (tid & 15) << 2;         // 16 col quads
    float acc[2][4] = {};
    for (int m = 0; m < 64; ++m) {
      float a0 = s2[r0 * 64 + m];
      float a1 = s2[(r0 + 1) * 64 + m];
#pragma unroll
      for (int j = 0; j < 4; ++j) {
        float vv = lQ[m * 64 + j0 + j];      // lQ holds V
        acc[0][j] += a0 * vv;
        acc[1][j] += a1 * vv;
      }
    }
    __syncthreads();                  // all old-K reads done before overwrite
#pragma unroll
    for (int r = 0; r < 2; ++r)
#pragma unroll
      for (int j = 0; j < 4; ++j)
        lK[(r0 + r) * 64 + j0 + j] = acc[r][j];
  }
  __syncthreads();

  // 7) out = s1 @ Kqv (64x64): 4 x 4 register tile; n = pw*64+ph
  {
    int r0 = (tid >> 4) << 2;         // 16 row quads
    int j0 = (tid & 15) << 2;         // 16 col quads
    float acc[4][4] = {};
    for (int m = 0; m < 32; ++m) {
      float a0 = s1[(r0 + 0) * 32 + m];
      float a1 = s1[(r0 + 1) * 32 + m];
      float a2 = s1[(r0 + 2) * 32 + m];
      float a3 = s1[(r0 + 3) * 32 + m];
#pragma unroll
      for (int j = 0; j < 4; ++j) {
        float kv = lK[m * 64 + j0 + j];      // lK holds Kqv
        acc[0][j] += a0 * kv;
        acc[1][j] += a1 * kv;
        acc[2][j] += a2 * kv;
        acc[3][j] += a3 * kv;
      }
    }
#pragma unroll
    for (int r = 0; r < 4; ++r)
#pragma unroll
      for (int j = 0; j < 4; ++j) {
        int n = (r0 + r) * 64 + j0 + j;
        out[((size_t)b * N_ + n) * C_ + c] = acc[r][j];
      }
  }
}

// ---------------------------------------------------------------------------
// Launch: convert -> WMMA projections (async-LDS staged) -> per-channel attn
// Workspace: x_bf16 (64 MB) | W_bf16 x3 (1.5 MB) | activations bf16 (192 MB)
// ---------------------------------------------------------------------------
extern "C" void kernel_launch(void* const* d_in, const int* in_sizes, int n_in,
                              void* d_out, int out_size, void* d_ws, size_t ws_size,
                              hipStream_t stream) {
  const float* x  = (const float*)d_in[0];
  const float* Wq = (const float*)d_in[1];
  const float* bq = (const float*)d_in[2];
  const float* Wk = (const float*)d_in[3];
  const float* bk = (const float*)d_in[4];
  const float* Wv = (const float*)d_in[5];
  const float* bv = (const float*)d_in[6];
  float* out = (float*)d_out;

  unsigned short* xb = (unsigned short*)d_ws;                     // B*N*C
  unsigned short* wb = xb + (size_t)B_ * N_ * C_;                 // 3*C*C
  unsigned short* sb = wb + (size_t)3 * C_ * C_;                  // 3*B*C*N

  // Phase 0: fp32 -> bf16
  f2bf_kernel<<<4096, 256, 0, stream>>>(x,  xb, B_ * N_ * C_);
  f2bf_kernel<<<256,  256, 0, stream>>>(Wq, wb + (size_t)0 * C_ * C_, C_ * C_);
  f2bf_kernel<<<256,  256, 0, stream>>>(Wk, wb + (size_t)1 * C_ * C_, C_ * C_);
  f2bf_kernel<<<256,  256, 0, stream>>>(Wv, wb + (size_t)2 * C_ * C_, C_ * C_);

  // Phase 1: WMMA projections + sigmoid.
  // blocks = 3 proj * 16 batch * 4 d-slabs * 64 n-tiles = 12288 (8 waves each)
  proj_kernel<<<12288, 256, 0, stream>>>(xb, wb, bq, bk, bv, sb);

  // Phase 2: one workgroup per (b, c)
  attn_kernel<<<B_ * C_, 256, 0, stream>>>(sb, out);
}
